// SelfAttention_85409719649061
// MI455X (gfx1250) — matrix operand order repaired and checked by reading, not compiled, not run on hardware
//
#include <hip/hip_runtime.h>

#define NB 4
#define SQ 2048
#define EMB 512
#define NH 8
#define HD 64

static constexpr size_t PROJ_ELEMS = (size_t)NB * NH * SQ * HD; // 4,194,304 bf16 elems per tensor

typedef __attribute__((ext_vector_type(16))) __bf16        v16bf;
typedef __attribute__((ext_vector_type(8)))  float         v8f;
typedef __attribute__((ext_vector_type(8)))  unsigned int  v8u;
typedef __attribute__((ext_vector_type(4)))  int           v4i;

#define AS1 __attribute__((address_space(1)))
#define AS3 __attribute__((address_space(3)))

__device__ __forceinline__ unsigned short f2bf(float f) {
  union { __bf16 b; unsigned short s; } c;
  c.b = (__bf16)f;                                  // native f32->bf16 cvt
  return c.s;
}

__device__ __forceinline__ v8f wmma32(v16bf a, v16bf b, v8f c) {
  return __builtin_amdgcn_wmma_f32_16x16x32_bf16(false, a, false, b, (short)0, c, false, false);
}

// A-operand (16x32 bf16): lane holds row M=lane%16; packed pairs at
// K = 2v + 16*half (+kbase).
__device__ __forceinline__ v16bf loadA(const unsigned short* row, int half, int kbase) {
  union { v8u u; v16bf b; } t;
#pragma unroll
  for (int v = 0; v < 8; ++v) {
    int pk = v * 2 + half * 16 + kbase;
    t.u[v] = *(const unsigned int*)(row + pk);
  }
  return t.b;
}

// B-operand (32x16 bf16): lane holds N=lane%16, K = e + 16*half -> 16 contiguous bf16.
__device__ __forceinline__ v16bf loadB16(const unsigned short* p) {
  union { v8u u; v16bf b; } t;
  t.u = *(const v8u*)p;
  return t.b;
}

// ---- CDNA5 async global->LDS (per-lane 16B; one instr moves 512B/wave) ----
__device__ __forceinline__ void async_cp16(const void* g, void* l) {
#if __has_builtin(__builtin_amdgcn_global_load_async_to_lds_b128)
  __builtin_amdgcn_global_load_async_to_lds_b128(
      (AS1 v4i*)(unsigned long long)g,
      (AS3 v4i*)(unsigned)(unsigned long long)l, 0, 0);
#else
  unsigned           lo = (unsigned)(unsigned long long)l;  // LDS byte offset
  unsigned long long ga = (unsigned long long)g;            // 64-bit global addr
  asm volatile("global_load_async_to_lds_b128 %0, %1, off" :: "v"(lo), "v"(ga) : "memory");
#endif
}
__device__ __forceinline__ void wait_async4() {
#if __has_builtin(__builtin_amdgcn_s_wait_asynccnt)
  __builtin_amdgcn_s_wait_asynccnt(4);
#else
  asm volatile("s_wait_asynccnt 0x4" ::: "memory");
#endif
}
__device__ __forceinline__ void wait_async0() {
#if __has_builtin(__builtin_amdgcn_s_wait_asynccnt)
  __builtin_amdgcn_s_wait_asynccnt(0);
#else
  asm volatile("s_wait_asynccnt 0x0" ::: "memory");
#endif
}

// ---------------------------------------------------------------------------
// Kernel 0: one-time weight conversion to bf16.
// Wv/Wk/Wq (DxD) -> transposed Wt[e][d] (B-operand contiguous in K-dim d).
// W_out (E x E)  -> elementwise bf16 (already contiguous in K-dim j).
// ---------------------------------------------------------------------------
__global__ __launch_bounds__(256) void convert_weights_kernel(
    const float* __restrict__ Wv, const float* __restrict__ Wk, const float* __restrict__ Wq,
    const float* __restrict__ Wout,
    unsigned short* __restrict__ Wvt, unsigned short* __restrict__ Wkt,
    unsigned short* __restrict__ Wqt, unsigned short* __restrict__ Wob)
{
  int tid = blockIdx.x * 256 + threadIdx.x;
  if (tid < HD * HD) {
    int e = tid >> 6, d = tid & 63;
    Wvt[e * HD + d] = f2bf(Wv[d * HD + e]);
    Wkt[e * HD + d] = f2bf(Wk[d * HD + e]);
    Wqt[e * HD + d] = f2bf(Wq[d * HD + e]);
  }
  for (int i = tid; i < EMB * EMB; i += gridDim.x * 256)
    Wob[i] = f2bf(Wout[i]);
}

// ---------------------------------------------------------------------------
// Kernel 1: per-head QKV projection (shared DxD weights), bf16 WMMA.
// Q,K -> bf16 [N,H,S,D]; V -> bf16 [N,H,D,S] (transposed for P*V B-operand).
// ---------------------------------------------------------------------------
__global__ __launch_bounds__(128) void qkv_proj_kernel(
    const float* __restrict__ xv, const float* __restrict__ xk, const float* __restrict__ xq,
    const unsigned short* __restrict__ Wvt, const unsigned short* __restrict__ Wkt,
    const unsigned short* __restrict__ Wqt,
    unsigned short* __restrict__ Vt, unsigned short* __restrict__ Kb, unsigned short* __restrict__ Qb)
{
  const int lane  = threadIdx.x & 31;
  const int wave  = threadIdx.x >> 5;
  const int tile  = blockIdx.x * 4 + wave;          // 4096 tiles
  const int stile = tile & (SQ / 16 - 1);
  const int h     = (tile >> 7) & (NH - 1);
  const int n     = tile >> 10;
  const int l16   = lane & 15, half = lane >> 4;
  const int s0    = stile * 16;
  const int srow  = s0 + l16;

  const float* xs[3]          = { xv, xk, xq };
  const unsigned short* wt[3] = { Wvt, Wkt, Wqt };
  const v8f zero8 = { 0.f,0.f,0.f,0.f,0.f,0.f,0.f,0.f };
  const int base_nh = n * NH + h;

#pragma unroll
  for (int mtx = 0; mtx < 3; ++mtx) {
    const float* X = xs[mtx];
    const unsigned short* W = wt[mtx];

    // A tiles: rows = s, K-dim = d (2 chunks of 32), fp32 -> bf16 (read-once data)
    v16bf a[2];
    const float* xrow = X + ((size_t)(n * SQ + srow)) * EMB + h * HD;
#pragma unroll
    for (int c = 0; c < 2; ++c) {
      union { v8u u; v16bf b; } t;
#pragma unroll
      for (int v = 0; v < 8; ++v) {
        int pk = v * 2 + half * 16 + c * 32;
        t.u[v] = (unsigned)f2bf(xrow[pk]) | ((unsigned)f2bf(xrow[pk + 1]) << 16);
      }
      a[c] = t.b;
    }

    v8f acc[4] = { zero8, zero8, zero8, zero8 };
#pragma unroll
    for (int c = 0; c < 2; ++c)
#pragma unroll
      for (int ec = 0; ec < 4; ++ec)
        acc[ec] = wmma32(a[c], loadB16(W + (size_t)(ec * 16 + l16) * HD + c * 32 + half * 16), acc[ec]);

#pragma unroll
    for (int ec = 0; ec < 4; ++ec)
#pragma unroll
      for (int r = 0; r < 8; ++r) {
        int row = s0 + r + 8 * half;
        int col = ec * 16 + l16;
        unsigned short bv = f2bf(acc[ec][r]);
        if (mtx == 0)      Vt[((size_t)(base_nh * HD + col)) * SQ + row] = bv;
        else if (mtx == 1) Kb[((size_t)(base_nh * SQ + row)) * HD + col] = bv;
        else               Qb[((size_t)(base_nh * SQ + row)) * HD + col] = bv;
      }
  }
}

// ---------------------------------------------------------------------------
// Kernel 2: block-cooperative flash attention.
// Block = (n, h, 64 q-rows) = 4 waves; K/V tiles (32 keys) staged to LDS with
// async loads, double-buffered; each wave owns a 16-row q-tile.
// ---------------------------------------------------------------------------
__global__ __launch_bounds__(128) void flash_attn_kernel(
    const unsigned short* __restrict__ Qb, const unsigned short* __restrict__ Kb,
    const unsigned short* __restrict__ Vt, const int* __restrict__ mask,
    unsigned short* __restrict__ Aout)
{
  __shared__ __align__(32) unsigned short Kt [2][32 * HD];  // [key][d]   8KB
  __shared__ __align__(32) unsigned short Vtl[2][HD * 32];  // [d][key]   8KB
  __shared__ __align__(32) unsigned short Plds[4][16 * 32]; // per-wave P 4KB

  const int lane = threadIdx.x & 31;
  const int wave = threadIdx.x >> 5;
  const int bid  = blockIdx.x;                      // 1024 blocks
  const int qt   = bid & 31;
  const int h    = (bid >> 5) & 7;
  const int n    = bid >> 8;
  const int l16  = lane & 15, half = lane >> 4;
  const int s0   = qt * 64 + wave * 16;
  const int nh   = n * NH + h;

  const v8f zero8 = { 0.f,0.f,0.f,0.f,0.f,0.f,0.f,0.f };

  const unsigned short* kbase = Kb + (size_t)nh * SQ * HD;
  const unsigned short* vbase = Vt + (size_t)nh * HD * SQ;

  // Q as A-operand (16x64, two 32-K chunks), resident for whole loop
  const unsigned short* qrow = Qb + ((size_t)(nh * SQ + s0 + l16)) * HD;
  v16bf q0 = loadA(qrow, half, 0);
  v16bf q1 = loadA(qrow, half, 32);

  float mrun[8], lrun[8];
  v8f o[4] = { zero8, zero8, zero8, zero8 };
#pragma unroll
  for (int r = 0; r < 8; ++r) { mrun[r] = -3.0e38f; lrun[r] = 0.f; }

  const float scale = 0.044194173824159216f;        // 1/sqrt(512)
  const float L2E   = 1.4426950408889634f;

  const int* mrow[8];
#pragma unroll
  for (int r = 0; r < 8; ++r)
    mrow[r] = mask + ((size_t)n * SQ + (s0 + r + 8 * half)) * SQ + l16;

  // stage one 32-key tile: K rows are contiguous 4KB; V rows strided (64B each)
  auto issue_tile = [&](int buf, int k0) {
    const char* kg = (const char*)kbase + (size_t)k0 * (HD * 2);
    char*       kl = (char*)&Kt[buf][0];
#pragma unroll
    for (int j = 0; j < 2; ++j) {
      unsigned off = threadIdx.x * 16 + j * 2048;
      async_cp16(kg + off, kl + off);
    }
    const char* vg = (const char*)vbase;
    char*       vl = (char*)&Vtl[buf][0];
#pragma unroll
    for (int j = 0; j < 2; ++j) {
      int c = threadIdx.x + 128 * j;                // 256 x 16B chunks
      int d = c >> 2, w = (c & 3) * 16;
      async_cp16(vg + (size_t)d * (SQ * 2) + (size_t)k0 * 2 + w, vl + c * 16);
    }
  };

  issue_tile(0, 0);
  const int NIT = SQ / 32;
  for (int it = 0; it < NIT; ++it) {
    const int buf = it & 1;
    const int k0  = it * 32;
    if (it + 1 < NIT) { issue_tile(buf ^ 1, k0 + 32); wait_async4(); }
    else              { wait_async0(); }
    __syncthreads();                                // tile visible to all waves

    // scores: two 16x16 C tiles over K-dim d=64
    v8f s0v = zero8, s1v = zero8;
    {
      const unsigned short* kr0 = &Kt[buf][(size_t)l16 * HD + half * 16];
      const unsigned short* kr1 = &Kt[buf][(size_t)(16 + l16) * HD + half * 16];
      s0v = wmma32(q0, loadB16(kr0),      s0v);
      s0v = wmma32(q1, loadB16(kr0 + 32), s0v);
      s1v = wmma32(q0, loadB16(kr1),      s1v);
      s1v = wmma32(q1, loadB16(kr1 + 32), s1v);
    }

    // mask + scale
    float sm0[8], sm1[8];
#pragma unroll
    for (int r = 0; r < 8; ++r) {
      sm0[r] = (mrow[r][k0]      == 0) ? -1.0e30f : s0v[r] * scale;
      sm1[r] = (mrow[r][k0 + 16] == 0) ? -1.0e30f : s1v[r] * scale;
    }

    // online softmax; a row spans the 16 lanes of one half
#pragma unroll
    for (int r = 0; r < 8; ++r) {
      float t = fmaxf(sm0[r], sm1[r]);
#pragma unroll
      for (int off = 1; off < 16; off <<= 1) t = fmaxf(t, __shfl_xor(t, off, 32));
      float mnew = fmaxf(mrun[r], t);
      float corr = exp2f((mrun[r] - mnew) * L2E);
      mrun[r] = mnew;
      float p0 = exp2f((sm0[r] - mnew) * L2E);
      float p1 = exp2f((sm1[r] - mnew) * L2E);
      float rs = p0 + p1;
#pragma unroll
      for (int off = 1; off < 16; off <<= 1) rs += __shfl_xor(rs, off, 32);
      lrun[r] = lrun[r] * corr + rs;
      o[0][r] *= corr; o[1][r] *= corr; o[2][r] *= corr; o[3][r] *= corr;
      int row = r + 8 * half;
      Plds[wave][row * 32 + l16]      = f2bf(p0);
      Plds[wave][row * 32 + 16 + l16] = f2bf(p1);
    }

    // per-wave DS RAW on private P buffer: DS is in-order, just drain DScnt
    asm volatile("s_wait_dscnt 0x0" ::: "memory");
    v16bf pA = loadA(&Plds[wave][l16 * 32], half, 0);

    // O += P * V  from LDS V tile [d][key]
#pragma unroll
    for (int dc = 0; dc < 4; ++dc)
      o[dc] = wmma32(pA, loadB16(&Vtl[buf][(size_t)(dc * 16 + l16) * 32 + half * 16]), o[dc]);

    __syncthreads();                                // all waves done with buf
  }

  // normalize and store bf16 [N,S,H*D]
#pragma unroll
  for (int r = 0; r < 8; ++r) {
    float inv = 1.0f / fmaxf(lrun[r], 1e-20f);
    int row = s0 + r + 8 * half;
#pragma unroll
    for (int dc = 0; dc < 4; ++dc)
      Aout[((size_t)(n * SQ + row)) * EMB + h * HD + dc * 16 + l16] = f2bf(o[dc][r] * inv);
  }
}

// ---------------------------------------------------------------------------
// Kernel 3: out = A(8192x512 bf16) * W_out^T (bf16) + B_out  -> f32
// ---------------------------------------------------------------------------
__global__ __launch_bounds__(128) void out_proj_kernel(
    const unsigned short* __restrict__ Aout, const unsigned short* __restrict__ Wob,
    const float* __restrict__ Bout, float* __restrict__ Out)
{
  const int lane  = threadIdx.x & 31;
  const int wave  = threadIdx.x >> 5;
  const int tile  = blockIdx.x * 4 + wave;          // 4096 tiles
  const int mtile = tile & 511;
  const int etile = tile >> 9;                      // 0..7
  const int l16   = lane & 15, half = lane >> 4;
  const int r0    = mtile * 16, e0 = etile * 64;

  const v8f zero8 = { 0.f,0.f,0.f,0.f,0.f,0.f,0.f,0.f };
  const unsigned short* arow = Aout + (size_t)(r0 + l16) * EMB;

  v8f acc[4] = { zero8, zero8, zero8, zero8 };
  for (int j0 = 0; j0 < EMB; j0 += 32) {
    v16bf a = loadA(arow + j0, half, 0);
#pragma unroll
    for (int dc = 0; dc < 4; ++dc) {
      int eN = e0 + dc * 16 + l16;
      acc[dc] = wmma32(a, loadB16(Wob + (size_t)eN * EMB + j0 + half * 16), acc[dc]);
    }
  }

#pragma unroll
  for (int dc = 0; dc < 4; ++dc) {
    int eN = e0 + dc * 16 + l16;
    float bb = Bout[eN];
#pragma unroll
    for (int r = 0; r < 8; ++r)
      Out[(size_t)(r0 + r + 8 * half) * EMB + eN] = acc[dc][r] + bb;
  }
}

// ---------------------------------------------------------------------------
extern "C" void kernel_launch(void* const* d_in, const int* in_sizes, int n_in,
                              void* d_out, int out_size, void* d_ws, size_t ws_size,
                              hipStream_t stream) {
  (void)in_sizes; (void)n_in; (void)out_size; (void)ws_size;
  const float* values = (const float*)d_in[0];
  const float* keys   = (const float*)d_in[1];
  const float* query  = (const float*)d_in[2];
  const float* Wv     = (const float*)d_in[3];
  const float* Wk     = (const float*)d_in[4];
  const float* Wq     = (const float*)d_in[5];
  const float* Wout   = (const float*)d_in[6];
  const float* Bout   = (const float*)d_in[7];
  const int*   mask   = (const int*)d_in[8];
  float* out = (float*)d_out;

  unsigned short* Qb  = (unsigned short*)d_ws;      // bf16 [N,H,S,D]
  unsigned short* Kb  = Qb + PROJ_ELEMS;            // bf16 [N,H,S,D]
  unsigned short* Vt  = Kb + PROJ_ELEMS;            // bf16 [N,H,D,S]
  unsigned short* Ao  = Vt + PROJ_ELEMS;            // bf16 [N,S,H*D]
  unsigned short* Wvt = Ao + PROJ_ELEMS;            // bf16 [64,64] (transposed)
  unsigned short* Wkt = Wvt + HD * HD;
  unsigned short* Wqt = Wkt + HD * HD;
  unsigned short* Wob = Wqt + HD * HD;              // bf16 [512,512]

  dim3 blk128(128);
  convert_weights_kernel<<<64, 256, 0, stream>>>(Wv, Wk, Wq, Wout, Wvt, Wkt, Wqt, Wob);
  qkv_proj_kernel  <<<1024, blk128, 0, stream>>>(values, keys, query, Wvt, Wkt, Wqt, Vt, Kb, Qb);
  flash_attn_kernel<<<1024, blk128, 0, stream>>>(Qb, Kb, Vt, mask, Ao);
  out_proj_kernel  <<<1024, blk128, 0, stream>>>(Ao, Wob, Bout, out);
}
